// Model_41704132444967
// MI455X (gfx1250) — matrix operand order repaired and checked by reading, not compiled, not run on hardware
//
#include <hip/hip_runtime.h>

// ---------------- types for WMMA ----------------
typedef __attribute__((ext_vector_type(16))) __bf16 v16bf;
typedef __attribute__((ext_vector_type(8)))  float  v8f;

#define D_MODEL 640
#define SEQ     33
#define BATCH   4
#define ROWS    (BATCH*SEQ)   // 132
#define ROWS_PAD 144
#define NHEAD   8
#define DHEAD   80
#define FFD     2560
#define VOCAB   30000
#define NLAYER  6
#define QKV_N   (3*D_MODEL)   // 1920
#define BOS_TOK 1
#define EOS_TOK 2

// ---------------------------------------------------------------------------
// Pack an fp32 KxN weight matrix into bf16 WMMA B-fragment order.
// Tile (kt,nt): lane l holds 16 contiguous bf16 = B[kt*32 + (l/16)*16 + e][nt*16 + l%16]
// flat index = ((nt*Ktiles + kt)*32 + l)*16 + e
// ---------------------------------------------------------------------------
__global__ void pack_w_kernel(const float* __restrict__ W, __bf16* __restrict__ out,
                              int K, int N) {
    long long idx = (long long)blockIdx.x * blockDim.x + threadIdx.x;
    long long total = (long long)K * N;
    if (idx >= total) return;
    int Ktiles = K >> 5;
    int e = (int)(idx & 15);
    int l = (int)((idx >> 4) & 31);
    long long t = idx >> 9;
    int kt = (int)(t % Ktiles);
    int nt = (int)(t / Ktiles);
    int kk = kt * 32 + ((l >> 4) << 4) + e;
    int n  = nt * 16 + (l & 15);
    out[idx] = (__bf16)W[(size_t)kk * N + n];
}

// ---------------------------------------------------------------------------
// Pack fp32 activations [M_PAD x K] into bf16 WMMA A-fragment order.
// A layout: row = lane%16, K(e) = (lane/16)*16 + e (matches B packing)
// flat index = ((mt*Ktiles + kt)*32 + l)*16 + e  -> lane loads 32B contiguous.
// ---------------------------------------------------------------------------
__global__ void pack_a_kernel(const float* __restrict__ A, __bf16* __restrict__ out,
                              int K, long long total) {
    long long idx = (long long)blockIdx.x * blockDim.x + threadIdx.x;
    if (idx >= total) return;
    int Ktiles = K >> 5;
    int e = (int)(idx & 15);
    int l = (int)((idx >> 4) & 31);
    long long t = idx >> 9;
    int kt = (int)(t % Ktiles);
    int mt = (int)(t / Ktiles);
    int row = mt * 16 + (l & 15);
    int kk  = kt * 32 + ((l >> 4) << 4) + e;
    out[idx] = (__bf16)A[(size_t)row * K + kk];
}

__global__ void concat_bias_kernel(const float* bq, const float* bk, const float* bv,
                                   float* out) {
    int i = blockIdx.x * blockDim.x + threadIdx.x;
    if (i < D_MODEL) {
        out[i]             = bq[i];
        out[D_MODEL + i]   = bk[i];
        out[2*D_MODEL + i] = bv[i];
    }
}

// ---------------------------------------------------------------------------
// Sequence init: inp[b][0]=BOS, inp[b][1..8]=initial words, rest 0. active=1.
// ---------------------------------------------------------------------------
__global__ void init_kernel(const int* __restrict__ iw, int* __restrict__ inp,
                            int* __restrict__ active) {
    int t = threadIdx.x;
    if (t < ROWS) {
        int b = t / SEQ, s = t % SEQ;
        int v = 0;
        if (s == 0) v = BOS_TOK;
        else if (s <= 8) v = iw[b * 8 + s - 1];
        inp[t] = v;
    }
    if (t == 0) active[0] = 1;
}

__global__ void embed_kernel(const float* __restrict__ emb, const int* __restrict__ inp,
                             float* __restrict__ h) {
    int idx = blockIdx.x * blockDim.x + threadIdx.x;
    if (idx < ROWS * D_MODEL) {
        int r = idx / D_MODEL, d = idx % D_MODEL;
        h[idx] = emb[(size_t)inp[r] * D_MODEL + d];
    }
}

// ---------------------------------------------------------------------------
// WMMA GEMM, both operands pre-packed bf16 fragments.
// One wave per block; each wave computes MT vertically-stacked 16x16 tiles
// sharing one B fragment per K-step (cuts B L2 traffic by MT).
// Addresses computed by direct indexing off the __restrict__ kernarg bases so
// the loads stay in the GLOBAL address space (global_load_b128, LOADcnt only).
// grid = (N/16, Mtiles/MT), block = 32.
// ---------------------------------------------------------------------------
template <int MT>
__global__ void gemm_wmma_kernel(const __bf16* __restrict__ Ap,
                                 const __bf16* __restrict__ Bp,
                                 const float* __restrict__ bias,
                                 float* __restrict__ C,
                                 int M, int K, int N, int relu) {
    int lane = threadIdx.x;          // 0..31
    int nt   = blockIdx.x;
    int mtb  = blockIdx.y * MT;
    int Ktiles = K >> 5;
    int g = lane >> 4;

    v8f acc[MT] = {};
    size_t boff = ((size_t)nt * Ktiles) * 512 + (size_t)lane * 16;
    size_t aoff0 = ((size_t)mtb * Ktiles) * 512 + (size_t)lane * 16;
    size_t astride = (size_t)Ktiles * 512;   // between consecutive M tiles

    for (int kt = 0; kt < Ktiles; ++kt) {
        v16bf b = *(const v16bf*)(Bp + boff);
        if (kt + 8 < Ktiles)
            __builtin_prefetch(Bp + boff + 512 * 8, 0, 1);  // global_prefetch_b8
        boff += 512;
#pragma unroll
        for (int i = 0; i < MT; ++i) {
            v16bf a = *(const v16bf*)(Ap + aoff0 + (size_t)i * astride);
            acc[i] = __builtin_amdgcn_wmma_f32_16x16x32_bf16(
                /*neg_a=*/false, a, /*neg_b=*/false, b,
                /*c_mod=*/(short)0, acc[i], /*reuse_a=*/false, /*reuse_b=*/false);
        }
        aoff0 += 512;
    }

    int col = nt * 16 + (lane & 15);
    float bv = bias[col];
#pragma unroll
    for (int i = 0; i < MT; ++i) {
#pragma unroll
        for (int r = 0; r < 8; ++r) {
            int m = (mtb + i) * 16 + (g << 3) + r;
            if (m < M) {
                float v = acc[i][r] + bv;
                if (relu) v = fmaxf(v, 0.f);
                C[(size_t)m * N + col] = v;
            }
        }
    }
}

// ---------------------------------------------------------------------------
// Attention: block per (b, head); thread sq handles one query row.
// qkv layout: row-major [ROWS_PAD x 1920], q|k|v at col offsets 0/640/1280.
// mask: causal (sk>sq) OR pad (sk >= 32)
// ---------------------------------------------------------------------------
__global__ void attn_kernel(const float* __restrict__ qkv, float* __restrict__ o) {
    int b  = blockIdx.x >> 3;
    int hh = blockIdx.x & 7;
    int sq = threadIdx.x;
    if (sq >= SEQ) return;
    const float scale = 0.11180339887498948f;  // 1/sqrt(80)
    int qrow = b * SEQ + sq;
    const float* qp = qkv + (size_t)qrow * QKV_N + hh * DHEAD;

    float sc[SEQ];
    float mx = -1e30f;
    for (int sk = 0; sk < SEQ; ++sk) {
        const float* kp = qkv + (size_t)(b * SEQ + sk) * QKV_N + D_MODEL + hh * DHEAD;
        float d = 0.f;
        for (int e = 0; e < DHEAD; ++e) d += qp[e] * kp[e];
        d *= scale;
        if (sk > sq || sk >= SEQ - 1) d -= 1e9f;
        sc[sk] = d;
        mx = fmaxf(mx, d);
    }
    float sum = 0.f;
    for (int sk = 0; sk < SEQ; ++sk) { sc[sk] = __expf(sc[sk] - mx); sum += sc[sk]; }
    float inv = 1.f / sum;

    float accv[DHEAD];
    for (int e = 0; e < DHEAD; ++e) accv[e] = 0.f;
    for (int sk = 0; sk < SEQ; ++sk) {
        float a = sc[sk] * inv;
        const float* vp = qkv + (size_t)(b * SEQ + sk) * QKV_N + 2 * D_MODEL + hh * DHEAD;
        for (int e = 0; e < DHEAD; ++e) accv[e] += a * vp[e];
    }
    float* op = o + (size_t)qrow * D_MODEL + hh * DHEAD;
    for (int e = 0; e < DHEAD; ++e) op[e] = accv[e];
}

// ---------------------------------------------------------------------------
// Post-LN residual: h = LN(h + delta) * gamma + beta, one block per row
// ---------------------------------------------------------------------------
__global__ void ln_kernel(float* __restrict__ h, const float* __restrict__ delta,
                          const float* __restrict__ gamma, const float* __restrict__ beta) {
    __shared__ float xs[D_MODEL];
    __shared__ float ssum[256], ssq[256];
    int row = blockIdx.x;
    int tid = threadIdx.x;
    float lsum = 0.f, lsq = 0.f;
    for (int i = tid; i < D_MODEL; i += 256) {
        float x = h[(size_t)row * D_MODEL + i] + delta[(size_t)row * D_MODEL + i];
        xs[i] = x; lsum += x; lsq += x * x;
    }
    ssum[tid] = lsum; ssq[tid] = lsq;
    __syncthreads();
    for (int st = 128; st > 0; st >>= 1) {
        if (tid < st) { ssum[tid] += ssum[tid + st]; ssq[tid] += ssq[tid + st]; }
        __syncthreads();
    }
    float mean = ssum[0] * (1.f / D_MODEL);
    float var  = ssq[0] * (1.f / D_MODEL) - mean * mean;
    float inv  = rsqrtf(var + 1e-6f);
    for (int i = tid; i < D_MODEL; i += 256)
        h[(size_t)row * D_MODEL + i] = (xs[i] - mean) * inv * gamma[i] + beta[i];
}

// Gather the 4 rows at position cl into a zero-padded 16 x 640 A tile
__global__ void gather_kernel(const float* __restrict__ h, float* __restrict__ lmA, int cl) {
    int idx = blockIdx.x * blockDim.x + threadIdx.x;
    if (idx < 16 * D_MODEL) {
        int r = idx / D_MODEL, d = idx % D_MODEL;
        lmA[idx] = (r < BATCH) ? h[(size_t)(r * SEQ + cl) * D_MODEL + d] : 0.f;
    }
}

// argmax over vocab, first-index tie-break; block per batch
__global__ void argmax_kernel(const float* __restrict__ logits, int* __restrict__ toks) {
    __shared__ float bv[256];
    __shared__ int   bidx[256];
    int b = blockIdx.x, tid = threadIdx.x;
    const float* lp = logits + (size_t)b * VOCAB;
    float best = -1e30f; int bi = 0x7fffffff;
    for (int i = tid; i < VOCAB; i += 256) {
        float v = lp[i];
        if (v > best || (v == best && i < bi)) { best = v; bi = i; }
    }
    bv[tid] = best; bidx[tid] = bi;
    __syncthreads();
    for (int st = 128; st > 0; st >>= 1) {
        if (tid < st) {
            if (bv[tid + st] > bv[tid] ||
                (bv[tid + st] == bv[tid] && bidx[tid + st] < bidx[tid])) {
                bv[tid] = bv[tid + st]; bidx[tid] = bidx[tid + st];
            }
        }
        __syncthreads();
    }
    if (tid == 0) toks[b] = bidx[0];
}

// Emulates while-loop: only write tokens while active; stop when all EOS.
__global__ void update_kernel(int* __restrict__ inp, const int* __restrict__ toks,
                              int* __restrict__ active, int cl) {
    if (threadIdx.x == 0 && blockIdx.x == 0) {
        if (active[0]) {
            int alleos = 1;
            for (int b = 0; b < BATCH; ++b) {
                inp[b * SEQ + cl + 1] = toks[b];
                if (toks[b] != EOS_TOK) alleos = 0;
            }
            if (alleos) active[0] = 0;
        }
    }
}

__global__ void out_kernel(const int* __restrict__ inp, int* __restrict__ out) {
    int i = threadIdx.x;
    if (i < BATCH * 32) {
        int b = i >> 5, s = i & 31;
        out[i] = inp[b * SEQ + 1 + s];
    }
}

// ---------------------------------------------------------------------------
extern "C" void kernel_launch(void* const* d_in, const int* in_sizes, int n_in,
                              void* d_out, int out_size, void* d_ws, size_t ws_size,
                              hipStream_t stream) {
    const int*   iw   = (const int*)  d_in[0];
    const float* emb  = (const float*)d_in[2];
    const float* Wq   = (const float*)d_in[3];  const float* bq  = (const float*)d_in[4];
    const float* Wk   = (const float*)d_in[5];  const float* bk  = (const float*)d_in[6];
    const float* Wv   = (const float*)d_in[7];  const float* bv  = (const float*)d_in[8];
    const float* Wo   = (const float*)d_in[9];  const float* bo  = (const float*)d_in[10];
    const float* ln1s = (const float*)d_in[11]; const float* ln1b= (const float*)d_in[12];
    const float* W1   = (const float*)d_in[13]; const float* b1  = (const float*)d_in[14];
    const float* W2   = (const float*)d_in[15]; const float* b2  = (const float*)d_in[16];
    const float* ln2s = (const float*)d_in[17]; const float* ln2b= (const float*)d_in[18];
    const float* lmW  = (const float*)d_in[19]; const float* lmb = (const float*)d_in[20];

    // ---- carve workspace ----
    char* ws = (char*)d_ws;
    size_t off = 0;
    auto carve = [&](size_t bytes) -> char* {
        char* p = ws + off;
        off += (bytes + 255) & ~(size_t)255;
        return p;
    };
    // bf16 packed weights (L2-resident across decode steps: ~97 MB < 192 MB L2)
    __bf16* pQKV[NLAYER]; __bf16* pWo[NLAYER]; __bf16* pW1[NLAYER]; __bf16* pW2[NLAYER];
    for (int l = 0; l < NLAYER; ++l) {
        pQKV[l] = (__bf16*)carve((size_t)D_MODEL * QKV_N * 2);     // q|k|v fused
        pWo[l]  = (__bf16*)carve((size_t)D_MODEL * D_MODEL * 2);
        pW1[l]  = (__bf16*)carve((size_t)D_MODEL * FFD * 2);
        pW2[l]  = (__bf16*)carve((size_t)FFD * D_MODEL * 2);
    }
    __bf16* pLM = (__bf16*)carve((size_t)D_MODEL * VOCAB * 2);
    // bf16 packed activations (reused for every GEMM's A operand)
    __bf16* pA = (__bf16*)carve((size_t)ROWS_PAD * FFD * 2);
    // fp32 activations (padded rows stay zero via the memset below)
    float* qkv    = (float*)carve((size_t)ROWS_PAD * QKV_N * 4);
    float* ao     = (float*)carve((size_t)ROWS_PAD * D_MODEL * 4);
    float* tbuf   = (float*)carve((size_t)ROWS_PAD * D_MODEL * 4);
    float* fbuf   = (float*)carve((size_t)ROWS_PAD * FFD * 4);
    float* h      = (float*)carve((size_t)ROWS_PAD * D_MODEL * 4);
    float* lmA    = (float*)carve((size_t)16 * D_MODEL * 4);
    float* logits = (float*)carve((size_t)16 * VOCAB * 4);
    float* bqkv   = (float*)carve((size_t)NLAYER * QKV_N * 4);
    int*   inp    = (int*)carve(ROWS * 4);
    int*   toks   = (int*)carve(BATCH * 4);
    int*   active = (int*)carve(4);

    // zero everything (pad rows of activation buffers must be finite/zero)
    hipMemsetAsync(d_ws, 0, off <= ws_size ? off : ws_size, stream);

    // ---- one-time weight packing (every call; deterministic) ----
    auto pack = [&](const float* src, __bf16* dst, int K, int N) {
        long long total = (long long)K * N;
        pack_w_kernel<<<(int)((total + 255) / 256), 256, 0, stream>>>(src, dst, K, N);
    };
    for (int l = 0; l < NLAYER; ++l) {
        size_t wsz = (size_t)D_MODEL * D_MODEL;
        pack(Wq + l * wsz, pQKV[l],                 D_MODEL, D_MODEL);
        pack(Wk + l * wsz, pQKV[l] + wsz,           D_MODEL, D_MODEL);
        pack(Wv + l * wsz, pQKV[l] + 2 * wsz,       D_MODEL, D_MODEL);
        pack(Wo + l * wsz, pWo[l],                  D_MODEL, D_MODEL);
        pack(W1 + (size_t)l * D_MODEL * FFD, pW1[l], D_MODEL, FFD);
        pack(W2 + (size_t)l * FFD * D_MODEL, pW2[l], FFD, D_MODEL);
        concat_bias_kernel<<<3, 256, 0, stream>>>(bq + l * D_MODEL, bk + l * D_MODEL,
                                                  bv + l * D_MODEL, bqkv + l * QKV_N);
    }
    pack(lmW, pLM, D_MODEL, VOCAB);

    init_kernel<<<1, 160, 0, stream>>>(iw, inp, active);

    auto packA = [&](const float* src, int mpad, int K) {
        long long total = (long long)mpad * K;
        pack_a_kernel<<<(int)((total + 255) / 256), 256, 0, stream>>>(src, pA, K, total);
    };

    // ---- 24 greedy decode steps (cl = 8..31) ----
    for (int step = 0; step < 24; ++step) {
        int cl = 8 + step;
        embed_kernel<<<(ROWS * D_MODEL + 255) / 256, 256, 0, stream>>>(emb, inp, h);
        for (int l = 0; l < NLAYER; ++l) {
            // fused QKV: 132x640 @ 640x1920
            packA(h, ROWS_PAD, D_MODEL);
            gemm_wmma_kernel<3><<<dim3(QKV_N / 16, 3), 32, 0, stream>>>(
                pA, pQKV[l], bqkv + l * QKV_N, qkv, ROWS, D_MODEL, QKV_N, 0);
            attn_kernel<<<BATCH * NHEAD, 64, 0, stream>>>(qkv, ao);
            packA(ao, ROWS_PAD, D_MODEL);
            gemm_wmma_kernel<3><<<dim3(D_MODEL / 16, 3), 32, 0, stream>>>(
                pA, pWo[l], bo + l * D_MODEL, tbuf, ROWS, D_MODEL, D_MODEL, 0);
            ln_kernel<<<ROWS, 256, 0, stream>>>(h, tbuf, ln1s + l * D_MODEL, ln1b + l * D_MODEL);
            packA(h, ROWS_PAD, D_MODEL);
            gemm_wmma_kernel<3><<<dim3(FFD / 16, 3), 32, 0, stream>>>(
                pA, pW1[l], b1 + l * FFD, fbuf, ROWS, D_MODEL, FFD, 1);
            packA(fbuf, ROWS_PAD, FFD);
            gemm_wmma_kernel<3><<<dim3(D_MODEL / 16, 3), 32, 0, stream>>>(
                pA, pW2[l], b2 + l * D_MODEL, tbuf, ROWS, FFD, D_MODEL, 0);
            ln_kernel<<<ROWS, 256, 0, stream>>>(h, tbuf, ln2s + l * D_MODEL, ln2b + l * D_MODEL);
        }
        // LM head only for the 4 rows at position cl (M padded to one 16-tile)
        gather_kernel<<<(16 * D_MODEL + 255) / 256, 256, 0, stream>>>(h, lmA, cl);
        packA(lmA, 16, D_MODEL);
        gemm_wmma_kernel<1><<<dim3(VOCAB / 16, 1), 32, 0, stream>>>(
            pA, pLM, lmb, logits, 16, D_MODEL, VOCAB, 0);
        argmax_kernel<<<BATCH, 256, 0, stream>>>(logits, toks);
        update_kernel<<<1, 1, 0, stream>>>(inp, toks, active, cl);
    }

    out_kernel<<<1, 128, 0, stream>>>(inp, (int*)d_out);
}
